// VoLNPlanner_60679297958312
// MI455X (gfx1250) — matrix operand order repaired and checked by reading, not compiled, not run on hardware
//
#include <hip/hip_runtime.h>
#include <hip/hip_bf16.h>
#include <math.h>

// ---------------- problem constants ----------------
#define Bsz 32
#define Tt  16
#define Ee  768
#define Hh  1024
#define NHh 16
#define HDd 64
#define Ll  6
#define Rr  8
#define Kk  16
#define Nn  100000
#define Pp  9
#define HORh 9
#define Ss  37                  // 1 + 16 + 3 + 16 + 1
#define MTOK (Bsz * Ss)         // 1184
#define NEG_INF (-3.402823466e38f)

typedef __bf16 bf16_t;
typedef bf16_t v16bh __attribute__((ext_vector_type(16)));
typedef float  v8f   __attribute__((ext_vector_type(8)));
typedef unsigned int uint4v __attribute__((ext_vector_type(4)));

// pack two fp32 into one dword of 2x bf16 (round-to-nearest-even)
__device__ __forceinline__ unsigned pack2bf(float f0, float f1) {
  unsigned u0 = __builtin_bit_cast(unsigned, f0);
  unsigned u1 = __builtin_bit_cast(unsigned, f1);
  u0 = u0 + 0x7FFFu + ((u0 >> 16) & 1u);
  u1 = u1 + 0x7FFFu + ((u1 >> 16) & 1u);
  return (u0 >> 16) | (u1 & 0xFFFF0000u);
}

// load 16 contiguous bf16 (8 dwords, 16B-aligned) from LDS as a WMMA operand
__device__ __forceinline__ v16bh ld_op16(const unsigned* p) {
  union { uint4v q[2]; v16bh v; } u;
  u.q[0] = *(const uint4v*)(p);
  u.q[1] = *(const uint4v*)(p + 4);
  return u.v;
}

__device__ __forceinline__ float gelu_f(float x) {
  return 0.5f * x * (1.0f + erff(x * 0.70710678118654752f));
}

// ---------------------------------------------------------------------------
// WMMA GEMM: C[M,N] = A[M,K] @ W[N,K]^T (+bias) (+loraT@loraB^T) (gelu?) (+res)
// A fp32 row-major (stride lda), W fp32 row-major N x K.
// Block: 256 threads (8 waves). Tile: 64M x 128N. Each wave: 16N strip,
// 4 M-subtiles of v_wmma_f32_16x16x32_bf16 with fp32 accumulation.
// Double-buffered LDS (bf16, natural k order, no permute) with
// register staging: tile k+1 global loads issue before tile k's WMMA chain.
//   A operand element e -> k = half*16 + e: natural order, no permute.
//   B operand element e -> k = half*16 + e: natural order, no permute.
// Row stride = 20 dwords (40 bf16): 80B, keeps 16B alignment, spreads banks.
// ---------------------------------------------------------------------------
#define ROWD 20   // dwords per LDS tile row

__global__ __launch_bounds__(256) void wmma_gemm_kernel(
    const float* __restrict__ A, int lda,
    const float* __restrict__ W,
    const float* __restrict__ bias,
    const float* __restrict__ loraT, const float* __restrict__ loraB,
    const float* __restrict__ residual,
    float* __restrict__ C, int ldc,
    int M, int N, int Kd, int flags)   // flags bit0 = gelu
{
  __shared__ __align__(16) unsigned sA[2][64 * ROWD];      // 2 x 5  KB
  __shared__ __align__(16) unsigned sB[2][8][16 * ROWD];   // 2 x 10 KB

  const int wave  = threadIdx.x >> 5;
  const int lane  = threadIdx.x & 31;
  const int tileM = blockIdx.y * 64;
  const int n0    = blockIdx.x * 128 + wave * 16;
  const int nn    = lane & 15;
  const int hf    = lane >> 4;

  const bool mfull = (tileM + 64 <= M);
  const bool nfull = (n0 + 16 <= N);
  const bool ldaEven = ((lda & 1) == 0);
  const bool kwEven  = ((Kd & 1) == 0);

  // register staging: A = 4 dwords/thread (64x16 dwords / 256), B = 8 dwords/lane
  float fa[4][2];
  float fb[8][2];

  auto fetchA = [&](int k0) {
    if (mfull && (k0 + 32 <= Kd)) {
      if (ldaEven) {
#pragma unroll
        for (int j = 0; j < 4; ++j) {
          int i = (int)threadIdx.x + j * 256;
          int m = i >> 4, kd = i & 15;
          const float2* p = (const float2*)(A + (size_t)(tileM + m) * lda + k0 + 2 * kd);
          float2 f = *p;
          fa[j][0] = f.x; fa[j][1] = f.y;
        }
      } else {
#pragma unroll
        for (int j = 0; j < 4; ++j) {
          int i = (int)threadIdx.x + j * 256;
          int m = i >> 4, kd = i & 15;
          const float* p = A + (size_t)(tileM + m) * lda + k0 + 2 * kd;
          fa[j][0] = p[0]; fa[j][1] = p[1];
        }
      }
    } else {
#pragma unroll
      for (int j = 0; j < 4; ++j) {
        int i = (int)threadIdx.x + j * 256;
        int m = i >> 4, kd = i & 15;
        int gm = tileM + m, gk0 = k0 + 2 * kd;
        const float* ap = A + (size_t)gm * lda;
        fa[j][0] = (gm < M && gk0     < Kd) ? ap[gk0]     : 0.0f;
        fa[j][1] = (gm < M && gk0 + 1 < Kd) ? ap[gk0 + 1] : 0.0f;
      }
    }
  };

  auto fetchB = [&](int k0) {
    if (nfull && (k0 + 32 <= Kd)) {
      if (kwEven) {
#pragma unroll
        for (int j = 0; j < 8; ++j) {
          int i = lane + j * 32;
          int n = i >> 4, kd = i & 15;
          const float2* p = (const float2*)(W + (size_t)(n0 + n) * Kd + k0 + 2 * kd);
          float2 f = *p;
          fb[j][0] = f.x; fb[j][1] = f.y;
        }
      } else {
#pragma unroll
        for (int j = 0; j < 8; ++j) {
          int i = lane + j * 32;
          int n = i >> 4, kd = i & 15;
          const float* p = W + (size_t)(n0 + n) * Kd + k0 + 2 * kd;
          fb[j][0] = p[0]; fb[j][1] = p[1];
        }
      }
    } else {
#pragma unroll
      for (int j = 0; j < 8; ++j) {
        int i = lane + j * 32;
        int n = i >> 4, kd = i & 15;
        int gn = n0 + n, gk0 = k0 + 2 * kd;
        const float* wpr = W + (size_t)gn * Kd;
        fb[j][0] = (gn < N && gk0     < Kd) ? wpr[gk0]     : 0.0f;
        fb[j][1] = (gn < N && gk0 + 1 < Kd) ? wpr[gk0 + 1] : 0.0f;
      }
    }
  };

  auto storeTiles = [&](int buf) {
#pragma unroll
    for (int j = 0; j < 4; ++j) {
      int i = (int)threadIdx.x + j * 256;
      int m = i >> 4, kd = i & 15;
      int pd = kd;   // natural k order (matches B)
      sA[buf][m * ROWD + pd] = pack2bf(fa[j][0], fa[j][1]);
    }
#pragma unroll
    for (int j = 0; j < 8; ++j) {
      int i = lane + j * 32;
      int n = i >> 4, kd = i & 15;
      sB[buf][wave][n * ROWD + kd] = pack2bf(fb[j][0], fb[j][1]);
    }
  };

  v8f acc[4];
#pragma unroll
  for (int t = 0; t < 4; ++t)
#pragma unroll
    for (int j = 0; j < 8; ++j) acc[t][j] = 0.0f;

  // prologue: stage first tile
  fetchA(0); fetchB(0); storeTiles(0);
  int buf = 0;

  for (int k0 = 0; k0 < Kd; k0 += 32) {
    const int k1 = k0 + 32;
    const bool more = (k1 < Kd);
    if (more) { fetchA(k1); fetchB(k1); }   // global loads in flight during WMMA
    __syncthreads();                         // current buffer tile visible

    v16bh b = ld_op16(&sB[buf][wave][nn * ROWD + hf * 8]);
    v16bh av[4];
#pragma unroll
    for (int mt = 0; mt < 4; ++mt)
      av[mt] = ld_op16(&sA[buf][(mt * 16 + nn) * ROWD + hf * 8]);
#pragma unroll
    for (int mt = 0; mt < 4; ++mt)
      acc[mt] = __builtin_amdgcn_wmma_f32_16x16x32_bf16(
          false, av[mt], false, b, (short)0, acc[mt], false, false);

    if (more) { storeTiles(buf ^ 1); buf ^= 1; }
  }

  // epilogue: C/D layout m = v + 8*half, n = lane&15
  const int ncol = n0 + nn;
  if (ncol < N) {
    float bval = bias ? bias[ncol] : 0.0f;
    float lb[8];
    if (loraT) {
#pragma unroll
      for (int r = 0; r < 8; ++r) lb[r] = loraB[(size_t)ncol * 8 + r];
    }
#pragma unroll
    for (int mt = 0; mt < 4; ++mt) {
#pragma unroll
      for (int v = 0; v < 8; ++v) {
        int gm = tileM + mt * 16 + hf * 8 + v;
        if (gm < M) {
          float val = acc[mt][v] + bval;
          if (loraT) {
#pragma unroll
            for (int r = 0; r < 8; ++r) val += loraT[(size_t)gm * 8 + r] * lb[r];
          }
          if (flags & 1) val = gelu_f(val);
          if (residual) val += residual[(size_t)gm * ldc + ncol];
          C[(size_t)gm * ldc + ncol] = val;
        }
      }
    }
  }
}

// ---------------------------------------------------------------------------
// top-K (K=16) per row over N=100000. One block per batch row, 256 threads.
// Per-thread sorted top-16, merged via 16 rounds of block argmax.
// ---------------------------------------------------------------------------
__global__ __launch_bounds__(256) void topk_kernel(
    float* __restrict__ sims, int* __restrict__ idx_out, float* __restrict__ sc_out)
{
  int b = blockIdx.x;
  float* row = sims + (size_t)b * Nn;

  float lv[16]; int li[16];
#pragma unroll
  for (int j = 0; j < 16; ++j) { lv[j] = NEG_INF; li[j] = 0; }
  for (int n = threadIdx.x; n < Nn; n += 256) {
    float v = row[n];
    if (v > lv[0]) {
      int p = 0;
      while (p < 15 && lv[p + 1] < v) { lv[p] = lv[p + 1]; li[p] = li[p + 1]; ++p; }
      lv[p] = v; li[p] = n;
    }
  }
  __shared__ float cv[4096];
  __shared__ int   ci[4096];
  __shared__ float rv[256];
  __shared__ int   ri[256];
  for (int j = 0; j < 16; ++j) { cv[threadIdx.x * 16 + j] = lv[j]; ci[threadIdx.x * 16 + j] = li[j]; }
  __syncthreads();

  for (int sel = 0; sel < 16; ++sel) {
    float best = NEG_INF; int bi = 0;
    for (int n = threadIdx.x; n < 4096; n += 256)
      if (cv[n] > best) { best = cv[n]; bi = n; }
    rv[threadIdx.x] = best; ri[threadIdx.x] = bi;
    __syncthreads();
    for (int s = 128; s > 0; s >>= 1) {
      if (threadIdx.x < s && rv[threadIdx.x + s] > rv[threadIdx.x]) {
        rv[threadIdx.x] = rv[threadIdx.x + s]; ri[threadIdx.x] = ri[threadIdx.x + s];
      }
      __syncthreads();
    }
    if (threadIdx.x == 0) {
      int w = ri[0];
      sc_out[b * 16 + sel]  = cv[w];
      idx_out[b * 16 + sel] = ci[w];
      cv[w] = NEG_INF;
    }
    __syncthreads();
  }
}

// ---------------------------------------------------------------------------
// attention: one block per (head, batch), S=37, HD=64
// ---------------------------------------------------------------------------
__global__ __launch_bounds__(128) void attention_kernel(
    const float* __restrict__ q, const float* __restrict__ k,
    const float* __restrict__ v, float* __restrict__ out)
{
  int h = blockIdx.x, b = blockIdx.y;
  __shared__ float sq[Ss * HDd], sk[Ss * HDd], sv[Ss * HDd], sp[Ss * Ss];
  const size_t base = (size_t)b * Ss * Hh + (size_t)h * HDd;
  for (int i = threadIdx.x; i < Ss * HDd; i += 128) {
    int s = i / HDd, d = i % HDd;
    size_t g = base + (size_t)s * Hh + d;
    sq[i] = q[g]; sk[i] = k[g]; sv[i] = v[g];
  }
  __syncthreads();
  for (int i = threadIdx.x; i < Ss * Ss; i += 128) {
    int s = i / Ss, t = i % Ss;
    float acc = 0.0f;
    for (int d = 0; d < HDd; ++d) acc += sq[s * HDd + d] * sk[t * HDd + d];
    sp[i] = acc * 0.125f;  // 1/sqrt(64)
  }
  __syncthreads();
  for (int s = threadIdx.x; s < Ss; s += 128) {
    float mx = NEG_INF;
    for (int t = 0; t < Ss; ++t) mx = fmaxf(mx, sp[s * Ss + t]);
    float sum = 0.0f;
    for (int t = 0; t < Ss; ++t) { float e = expf(sp[s * Ss + t] - mx); sp[s * Ss + t] = e; sum += e; }
    float inv = 1.0f / sum;
    for (int t = 0; t < Ss; ++t) sp[s * Ss + t] *= inv;
  }
  __syncthreads();
  for (int i = threadIdx.x; i < Ss * HDd; i += 128) {
    int s = i / HDd, d = i % HDd;
    float acc = 0.0f;
    for (int t = 0; t < Ss; ++t) acc += sp[s * Ss + t] * sv[t * HDd + d];
    out[base + (size_t)s * Hh + d] = acc;
  }
}

// ---------------------------------------------------------------------------
// layernorm: one block per row (D = 1024)
// ---------------------------------------------------------------------------
__global__ __launch_bounds__(256) void layernorm_kernel(
    const float* __restrict__ x, const float* __restrict__ g,
    const float* __restrict__ bb, float* __restrict__ y)
{
  const int D = Hh;
  int row = blockIdx.x;
  const float* xr = x + (size_t)row * D;
  __shared__ float red[256];
  float s = 0.0f;
  for (int i = threadIdx.x; i < D; i += 256) s += xr[i];
  red[threadIdx.x] = s; __syncthreads();
  for (int st = 128; st > 0; st >>= 1) {
    if (threadIdx.x < st) red[threadIdx.x] += red[threadIdx.x + st];
    __syncthreads();
  }
  float mean = red[0] / D;
  __syncthreads();
  float vs = 0.0f;
  for (int i = threadIdx.x; i < D; i += 256) { float d = xr[i] - mean; vs += d * d; }
  red[threadIdx.x] = vs; __syncthreads();
  for (int st = 128; st > 0; st >>= 1) {
    if (threadIdx.x < st) red[threadIdx.x] += red[threadIdx.x + st];
    __syncthreads();
  }
  float rstd = rsqrtf(red[0] / D + 1e-5f);
  for (int i = threadIdx.x; i < D; i += 256)
    y[(size_t)row * D + i] = (xr[i] - mean) * rstd * g[i] + bb[i];
}

// ---------------------------------------------------------------------------
// LoRA low-rank projection: t[m,r] = sum_k x[m,k] * Am[r,k]   (R = 8)
// ---------------------------------------------------------------------------
__global__ void lora_t_kernel(const float* __restrict__ x, int ldx,
                              const float* __restrict__ Am,
                              float* __restrict__ t, int M, int Kd)
{
  int i = blockIdx.x * blockDim.x + threadIdx.x;
  if (i >= M * 8) return;
  int m = i >> 3, r = i & 7;
  const float* xr = x + (size_t)m * ldx;
  const float* ar = Am + (size_t)r * Kd;
  float s = 0.0f;
  for (int kk = 0; kk < Kd; ++kk) s += xr[kk] * ar[kk];
  t[i] = s;
}

// ---------------------------------------------------------------------------
// small assembly kernels
// ---------------------------------------------------------------------------
__global__ void build_hist_in_kernel(const float* __restrict__ img,
                                     const float* __restrict__ hp,
                                     float* __restrict__ out)
{
  int i = blockIdx.x * blockDim.x + threadIdx.x;
  const int tot = Bsz * Tt * (Ee + Pp);
  if (i >= tot) return;
  int j = i % (Ee + Pp), bt = i / (Ee + Pp);
  out[i] = (j < Ee) ? img[(size_t)bt * Ee + j] : hp[(size_t)bt * Pp + (j - Ee)];
}

__global__ void build_imgs_in_kernel(const float* __restrict__ goal,
                                     const float* __restrict__ sub,
                                     const float* __restrict__ bea,
                                     const float* __restrict__ bank,
                                     const int* __restrict__ idx,
                                     const float* __restrict__ sc,
                                     float* __restrict__ out)
{
  int i = blockIdx.x * blockDim.x + threadIdx.x;
  const int tot = Bsz * 19 * Ee;
  if (i >= tot) return;
  int e = i % Ee; int tk = (i / Ee) % 19; int b = i / (19 * Ee);
  float v;
  if (tk == 0)      v = goal[(size_t)b * Ee + e];
  else if (tk == 1) v = sub [(size_t)b * Ee + e];
  else if (tk == 2) v = bea [(size_t)b * Ee + e];
  else {
    int j = tk - 3;
    v = bank[(size_t)idx[b * Kk + j] * Ee + e] * sc[b * Kk + j];
  }
  out[i] = v;
}

__global__ void assemble_x_kernel(const float* __restrict__ plan,
                                  const float* __restrict__ hist,
                                  const float* __restrict__ imgs,
                                  const float* __restrict__ prop,
                                  float* __restrict__ x)
{
  int i = blockIdx.x * blockDim.x + threadIdx.x;
  const int tot = MTOK * Hh;
  if (i >= tot) return;
  int col = i & (Hh - 1);
  int tok = (i >> 10) % Ss;
  int b   = i / (Ss * Hh);
  float v;
  if (tok == 0)       v = plan[col];
  else if (tok <= 16) v = hist[((size_t)b * Tt + (tok - 1)) * Hh + col];
  else if (tok <= 35) v = imgs[((size_t)b * 19 + (tok - 17)) * Hh + col];
  else                v = prop[(size_t)b * Hh + col];
  x[i] = v;
}

__global__ void build_ref_in_kernel(const float* __restrict__ xf,
                                    const float* __restrict__ anchor,
                                    float* __restrict__ out)
{
  int i = blockIdx.x * blockDim.x + threadIdx.x;
  const int tot = Bsz * (Hh + 3);
  if (i >= tot) return;
  int b = i / (Hh + 3), j = i % (Hh + 3);
  out[i] = (j < Hh) ? xf[(size_t)b * Ss * Hh + j] : anchor[b * 3 + (j - Hh)];
}

// ---------------------------------------------------------------------------
// host side
// ---------------------------------------------------------------------------
static inline void launch_gemm(hipStream_t s, const float* A, int lda, const float* W,
                               const float* bias, const float* loraT, const float* loraB,
                               const float* residual, float* C, int ldc,
                               int M, int N, int K, int flags) {
  dim3 grid((N + 127) / 128, (M + 63) / 64);
  wmma_gemm_kernel<<<grid, 256, 0, s>>>(A, lda, W, bias, loraT, loraB, residual,
                                        C, ldc, M, N, K, flags);
}

static inline int gsz(int n) { return (n + 255) / 256; }

extern "C" void kernel_launch(void* const* d_in, const int* in_sizes, int n_in,
                              void* d_out, int out_size, void* d_ws, size_t ws_size,
                              hipStream_t stream) {
  (void)in_sizes; (void)n_in; (void)out_size; (void)ws_size;
  const float* hist_img = (const float*)d_in[0];
  const float* cur_emb  = (const float*)d_in[1];
  const float* goal_emb = (const float*)d_in[2];
  const float* sub_emb  = (const float*)d_in[3];
  const float* bea_emb  = (const float*)d_in[4];
  const float* hist_pp  = (const float*)d_in[5];
  const float* proprio  = (const float*)d_in[6];
  const float* bank     = (const float*)d_in[7];
  const float* hp_W = (const float*)d_in[8];   const float* hp_b = (const float*)d_in[9];
  const float* ip_W = (const float*)d_in[10];  const float* ip_b = (const float*)d_in[11];
  const float* pp_W = (const float*)d_in[12];  const float* pp_b = (const float*)d_in[13];
  const float* plan_tok = (const float*)d_in[14];
  const float* ln1_g = (const float*)d_in[15]; const float* ln1_b = (const float*)d_in[16];
  const float* ln2_g = (const float*)d_in[17]; const float* ln2_b = (const float*)d_in[18];
  const float* qW = (const float*)d_in[19]; const float* qA = (const float*)d_in[20];
  const float* qB = (const float*)d_in[21]; const float* qb = (const float*)d_in[22];
  const float* kW = (const float*)d_in[23]; const float* kA = (const float*)d_in[24];
  const float* kB = (const float*)d_in[25]; const float* kb = (const float*)d_in[26];
  const float* vW = (const float*)d_in[27]; const float* vA = (const float*)d_in[28];
  const float* vB = (const float*)d_in[29]; const float* vb = (const float*)d_in[30];
  const float* oW = (const float*)d_in[31]; const float* oA = (const float*)d_in[32];
  const float* oB = (const float*)d_in[33]; const float* ob = (const float*)d_in[34];
  const float* f1W = (const float*)d_in[35]; const float* f1A = (const float*)d_in[36];
  const float* f1B = (const float*)d_in[37]; const float* f1b = (const float*)d_in[38];
  const float* f2W = (const float*)d_in[39]; const float* f2A = (const float*)d_in[40];
  const float* f2B = (const float*)d_in[41]; const float* f2b = (const float*)d_in[42];
  const float* fln_g = (const float*)d_in[43]; const float* fln_b = (const float*)d_in[44];
  const float* ah_W1 = (const float*)d_in[45]; const float* ah_b1 = (const float*)d_in[46];
  const float* ah_W2 = (const float*)d_in[47]; const float* ah_b2 = (const float*)d_in[48];
  const float* rh_W1 = (const float*)d_in[49]; const float* rh_b1 = (const float*)d_in[50];
  const float* rh_W2 = (const float*)d_in[51]; const float* rh_b2 = (const float*)d_in[52];
  const float* sh_W  = (const float*)d_in[53]; const float* sh_b  = (const float*)d_in[54];
  float* out = (float*)d_out;

  // bump-allocate workspace
  char* wp = (char*)d_ws;
  auto alloc = [&](size_t bytes) -> void* {
    void* r = (void*)wp;
    wp += (bytes + 255) & ~(size_t)255;
    return r;
  };
  float* sims     = (float*)alloc((size_t)Bsz * Nn * 4);
  int*   topk_i   = (int*)  alloc((size_t)Bsz * Kk * 4);
  float* topk_s   = (float*)alloc((size_t)Bsz * Kk * 4);
  float* hist_in  = (float*)alloc((size_t)Bsz * Tt * (Ee + Pp) * 4);
  float* hist_out = (float*)alloc((size_t)Bsz * Tt * Hh * 4);
  float* imgs_in  = (float*)alloc((size_t)Bsz * 19 * Ee * 4);
  float* imgs_out = (float*)alloc((size_t)Bsz * 19 * Hh * 4);
  float* prop_out = (float*)alloc((size_t)Bsz * Hh * 4);
  float* x        = (float*)alloc((size_t)MTOK * Hh * 4);
  float* hbuf     = (float*)alloc((size_t)MTOK * Hh * 4);
  float* qbuf     = (float*)alloc((size_t)MTOK * Hh * 4);
  float* kbuf     = (float*)alloc((size_t)MTOK * Hh * 4);
  float* vbuf     = (float*)alloc((size_t)MTOK * Hh * 4);
  float* abuf     = (float*)alloc((size_t)MTOK * Hh * 4);
  float* mid      = (float*)alloc((size_t)MTOK * 4 * Hh * 4);
  float* xf       = (float*)alloc((size_t)MTOK * Hh * 4);
  float* tbuf     = (float*)alloc((size_t)MTOK * Rr * 4);
  float* a1       = (float*)alloc((size_t)Bsz * Hh * 4);
  float* ref_in   = (float*)alloc((size_t)Bsz * (Hh + 3) * 4);
  float* r1       = (float*)alloc((size_t)Bsz * Hh * 4);

  // 1) retrieval: sims = current_emb @ bank^T   (WMMA, bank streamed once)
  launch_gemm(stream, cur_emb, Ee, bank, nullptr, nullptr, nullptr, nullptr,
              sims, Nn, Bsz, Nn, Ee, 0);
  topk_kernel<<<Bsz, 256, 0, stream>>>(sims, topk_i, topk_s);

  // 2) input projections
  build_hist_in_kernel<<<gsz(Bsz * Tt * (Ee + Pp)), 256, 0, stream>>>(hist_img, hist_pp, hist_in);
  build_imgs_in_kernel<<<gsz(Bsz * 19 * Ee), 256, 0, stream>>>(
      goal_emb, sub_emb, bea_emb, bank, topk_i, topk_s, imgs_in);
  launch_gemm(stream, hist_in, Ee + Pp, hp_W, hp_b, nullptr, nullptr, nullptr,
              hist_out, Hh, Bsz * Tt, Hh, Ee + Pp, 0);
  launch_gemm(stream, imgs_in, Ee, ip_W, ip_b, nullptr, nullptr, nullptr,
              imgs_out, Hh, Bsz * 19, Hh, Ee, 0);
  launch_gemm(stream, proprio, Pp, pp_W, pp_b, nullptr, nullptr, nullptr,
              prop_out, Hh, Bsz, Hh, Pp, 0);
  assemble_x_kernel<<<gsz(MTOK * Hh), 256, 0, stream>>>(plan_tok, hist_out, imgs_out, prop_out, x);

  // 3) transformer layers
  const size_t sW = (size_t)Hh * Hh, sA8 = (size_t)Rr * Hh, sB8 = (size_t)Hh * Rr;
  const size_t sW1 = (size_t)4 * Hh * Hh, sA1 = (size_t)Rr * Hh, sB1 = (size_t)4 * Hh * Rr;
  const size_t sA2 = (size_t)Rr * 4 * Hh, sB2 = (size_t)Hh * Rr;
  for (int l = 0; l < Ll; ++l) {
    layernorm_kernel<<<MTOK, 256, 0, stream>>>(x, ln1_g + (size_t)l * Hh, ln1_b + (size_t)l * Hh, hbuf);
    // Q
    lora_t_kernel<<<gsz(MTOK * Rr), 256, 0, stream>>>(hbuf, Hh, qA + l * sA8, tbuf, MTOK, Hh);
    launch_gemm(stream, hbuf, Hh, qW + l * sW, qb + (size_t)l * Hh, tbuf, qB + l * sB8,
                nullptr, qbuf, Hh, MTOK, Hh, Hh, 0);
    // K
    lora_t_kernel<<<gsz(MTOK * Rr), 256, 0, stream>>>(hbuf, Hh, kA + l * sA8, tbuf, MTOK, Hh);
    launch_gemm(stream, hbuf, Hh, kW + l * sW, kb + (size_t)l * Hh, tbuf, kB + l * sB8,
                nullptr, kbuf, Hh, MTOK, Hh, Hh, 0);
    // V
    lora_t_kernel<<<gsz(MTOK * Rr), 256, 0, stream>>>(hbuf, Hh, vA + l * sA8, tbuf, MTOK, Hh);
    launch_gemm(stream, hbuf, Hh, vW + l * sW, vb + (size_t)l * Hh, tbuf, vB + l * sB8,
                nullptr, vbuf, Hh, MTOK, Hh, Hh, 0);
    // attention
    attention_kernel<<<dim3(NHh, Bsz), 128, 0, stream>>>(qbuf, kbuf, vbuf, abuf);
    // O projection + residual into x
    lora_t_kernel<<<gsz(MTOK * Rr), 256, 0, stream>>>(abuf, Hh, oA + l * sA8, tbuf, MTOK, Hh);
    launch_gemm(stream, abuf, Hh, oW + l * sW, ob + (size_t)l * Hh, tbuf, oB + l * sB8,
                x, x, Hh, MTOK, Hh, Hh, 0);
    // MLP
    layernorm_kernel<<<MTOK, 256, 0, stream>>>(x, ln2_g + (size_t)l * Hh, ln2_b + (size_t)l * Hh, hbuf);
    lora_t_kernel<<<gsz(MTOK * Rr), 256, 0, stream>>>(hbuf, Hh, f1A + l * sA1, tbuf, MTOK, Hh);
    launch_gemm(stream, hbuf, Hh, f1W + l * sW1, f1b + (size_t)l * 4 * Hh, tbuf, f1B + l * sB1,
                nullptr, mid, 4 * Hh, MTOK, 4 * Hh, Hh, /*gelu*/1);
    lora_t_kernel<<<gsz(MTOK * Rr), 256, 0, stream>>>(mid, 4 * Hh, f2A + l * sA2, tbuf, MTOK, 4 * Hh);
    launch_gemm(stream, mid, 4 * Hh, f2W + l * sW1, f2b + (size_t)l * Hh, tbuf, f2B + l * sB2,
                x, x, Hh, MTOK, Hh, 4 * Hh, 0);
  }

  // 4) final LN + heads (ps = xf[:,0,:], row stride S*H)
  layernorm_kernel<<<MTOK, 256, 0, stream>>>(x, fln_g, fln_b, xf);
  // anchor head
  launch_gemm(stream, xf, Ss * Hh, ah_W1, ah_b1, nullptr, nullptr, nullptr,
              a1, Hh, Bsz, Hh, Hh, /*gelu*/1);
  launch_gemm(stream, a1, Hh, ah_W2, ah_b2, nullptr, nullptr, nullptr,
              out, 3, Bsz, 3, Hh, 0);                        // anchor -> out[0:96]
  // refinement head
  build_ref_in_kernel<<<gsz(Bsz * (Hh + 3)), 256, 0, stream>>>(xf, out, ref_in);
  launch_gemm(stream, ref_in, Hh + 3, rh_W1, rh_b1, nullptr, nullptr, nullptr,
              r1, Hh, Bsz, Hh, Hh + 3, /*gelu*/1);
  launch_gemm(stream, r1, Hh, rh_W2, rh_b2, nullptr, nullptr, nullptr,
              out + Bsz * 3, HORh * 3, Bsz, HORh * 3, Hh, 0); // waypoints -> out[96:960]
  // stop head
  launch_gemm(stream, xf, Ss * Hh, sh_W, sh_b, nullptr, nullptr, nullptr,
              out + Bsz * 3 + Bsz * HORh * 3, 1, Bsz, 1, Hh, 0); // stop -> out[960:992]
}
